// SGNCA_83468394430967
// MI455X (gfx1250) — compile-verified, matrix operand order checked by reading, not run on hardware
//
#include <hip/hip_runtime.h>
#include <hip/hip_bf16.h>
#include <cstdint>
#include <cstddef>

// ---------------- problem constants (match reference) ----------------
#define B_  16
#define S_  512
#define H_  512
#define NH_ 8
#define DH_ 64
#define FF_ 2048
#define MS_ (B_ * S_)          // 8192 rows
#define GE_ ((size_t)MS_ * H_) // 4,194,304 elements per [B,S,H] tensor

typedef __attribute__((ext_vector_type(16))) _Float16 v16h;
typedef __attribute__((ext_vector_type(8)))  _Float16 v8h;
typedef __attribute__((ext_vector_type(8)))  float    v8f;
typedef __attribute__((ext_vector_type(4)))  unsigned int v4u;
typedef __attribute__((ext_vector_type(8)))  int      v8i;
typedef __attribute__((ext_vector_type(4)))  int      v4i;

// Tensor Data Mover availability (device pass only; host pass sees 0)
#if defined(__has_builtin)
#  if __has_builtin(__builtin_amdgcn_tensor_load_to_lds) && \
      __has_builtin(__builtin_amdgcn_s_wait_tensorcnt)
#    define HAVE_TDM 1
#  endif
#endif
#ifndef HAVE_TDM
#  define HAVE_TDM 0
#endif

// B-tile staged in LDS with TDM padding: 32 f16 (16 dw) per row + 4 dw pad
// -> row stride 20 dwords = 40 f16 = 80 B (16-B aligned, conflict-free:
//    r*20 mod 64 distinct for r = 0..15).
#define BROW_ 40

// ---------------------------------------------------------------------
// WMMA 16x16x32 f16 fragment loader (CDNA5 ISA 16-bit A/B striping):
//   elems 0..7  <- K = half*8 + (0..7)       (contiguous 16 B)
//   elems 8..15 <- K = 16 + half*8 + (0..7)  (contiguous 16 B)
// Works for row-major A tiles and pre-transposed (N-major) B tiles.
// ---------------------------------------------------------------------
__device__ __forceinline__ v16h load_frag(const _Float16* p, int ld) {
  const int lane = threadIdx.x & 31;
  const int m    = lane & 15;
  const int hf   = lane >> 4;
  const _Float16* r = p + (size_t)m * ld + hf * 8;
  v8h lo = *(const v8h*)(r);
  v8h hi = *(const v8h*)(r + 16);
  return __builtin_shufflevector(lo, hi, 0,1,2,3,4,5,6,7,8,9,10,11,12,13,14,15);
}

#if HAVE_TDM
// 2-D TENSOR_LOAD_TO_LDS: tile_dim0=32 (K, contiguous), tile_dim1=64 rows,
// element = 2 B, LDS pad 4 dwords every 16 dwords (one row) -> BROW_ stride.
__device__ __forceinline__ void tdm_load_b_tile(unsigned lds_bytes,
                                                const _Float16* gptr, int K)
{
  const unsigned long long ga = (unsigned long long)(size_t)gptr;
  v4u g0;
  g0[0] = 1u;                                         // count=1, user D#
  g0[1] = lds_bytes;                                  // lds_addr
  g0[2] = (unsigned)(ga & 0xFFFFFFFFu);               // global_addr[31:0]
  g0[3] = (unsigned)((ga >> 32) & 0x01FFFFFFu) | (2u << 30); // [56:32] | type=2
  v8i g1;
  g1[0] = (1 << 16) | (1 << 20) | (3 << 22) | (3 << 25); // 2B, pad_en, ivl=16dw, amt=4dw
  g1[1] = (int)(((unsigned)K & 0xFFFFu) << 16);          // tensor_dim0[15:0]
  g1[2] = (int)((((unsigned)K >> 16) & 0xFFFFu) | (64u << 16)); // d0 hi | tensor_dim1 lo
  g1[3] = (int)(32u << 16);                              // tensor_dim1 hi | tile_dim0=32
  g1[4] = 64;                                            // tile_dim1=64, tile_dim2=0
  g1[5] = K;                                             // tensor_dim0_stride[31:0]
  g1[6] = 0;
  g1[7] = 0;
  v4i z4 = {0, 0, 0, 0};
#if defined(__clang_major__) && (__clang_major__ >= 23)
  v8i z8 = {0, 0, 0, 0, 0, 0, 0, 0};
  __builtin_amdgcn_tensor_load_to_lds(g0, g1, z4, z4, z8, 0);
#else
  __builtin_amdgcn_tensor_load_to_lds(g0, g1, z4, z4, 0);
#endif
}
#endif

// ---------------------------------------------------------------------
// GEMM: C[M,N] = A[M,K](f16) * WT[N,K](f16 pre-transposed) + bias, opt ReLU.
// 256 threads = 8 waves; wave owns a 16x16 C tile; block tile 32x64.
// B slab (64x32 f16) staged in LDS by the Tensor Data Mover, double-buffered;
// slab i+1 DMA overlaps slab i WMMAs. A slabs come straight from L2 as two
// global_load_b128 fragments per lane.
// ---------------------------------------------------------------------
__global__ __launch_bounds__(256) void gemm_wmma(
    const _Float16* __restrict__ A, const _Float16* __restrict__ WT,
    const float* __restrict__ bias, float* Cf, _Float16* Ch,
    int M, int N, int K, int relu)
{
  const int wave = threadIdx.x >> 5;
  const int lane = threadIdx.x & 31;
  const int ln   = lane & 15;
  const int hf   = lane >> 4;
  const int m0  = blockIdx.x * 32 + (wave & 1) * 16;
  const int bn0 = blockIdx.y * 64;
  const int n0  = bn0 + (wave >> 1) * 16;

  const _Float16* Ap = A + (size_t)m0 * K;
  v8f acc = {};

#if HAVE_TDM
  __shared__ __align__(16) _Float16 bsh[2][64 * BROW_];
  const _Float16* Bg = WT + (size_t)bn0 * K;   // block's B (weight) tile base
  if (wave == 0)
    tdm_load_b_tile((unsigned)(size_t)&bsh[0][0], Bg, K);
  if (wave == 0) __builtin_amdgcn_s_wait_tensorcnt((short)0);
  __syncthreads();

  int buf = 0;
  const _Float16* bloc = &bsh[0][0] + (size_t)((wave >> 1) * 16) * BROW_;
  for (int kb = 0; kb < K; kb += 32) {
    const bool more = (kb + 32) < K;
    if (more && wave == 0)                       // prefill next slab via TDM
      tdm_load_b_tile((unsigned)(size_t)&bsh[buf ^ 1][0], Bg + kb + 32, K);
    if (more) __builtin_prefetch(Ap + kb + 32, 0, 0);   // global_prefetch_b8

    v16h a = load_frag(Ap + kb, K);
    v16h b = load_frag(bloc + (size_t)buf * (64 * BROW_), BROW_);
    acc = __builtin_amdgcn_wmma_f32_16x16x32_f16(
        false, a, false, b, (short)0, acc, false, false);

    if (more) {
      if (wave == 0) __builtin_amdgcn_s_wait_tensorcnt((short)0);
      __syncthreads();
      buf ^= 1;
    }
  }
#else
  const _Float16* Bp = WT + (size_t)n0 * K;
  for (int kb = 0; kb < K; kb += 32) {
    if (kb + 32 < K) {
      __builtin_prefetch(Ap + kb + 32, 0, 0);
      __builtin_prefetch(Bp + kb + 32, 0, 0);
    }
    v16h a = load_frag(Ap + kb, K);
    v16h b = load_frag(Bp + kb, K);
    acc = __builtin_amdgcn_wmma_f32_16x16x32_f16(
        false, a, false, b, (short)0, acc, false, false);
  }
#endif

  const int n = n0 + ln;
  const float bv = bias ? bias[n] : 0.0f;
#pragma unroll
  for (int r = 0; r < 8; ++r) {
    float v = acc[r] + bv;
    if (relu) v = fmaxf(v, 0.0f);
    const size_t idx = (size_t)(m0 + r + 8 * hf) * N + n;
    if (Cf) Cf[idx] = v;
    if (Ch) Ch[idx] = (_Float16)v;
  }
}

// ---------------------------------------------------------------------
// Flash-style fused attention. Grid = B*NH*(S/64); block = 128 (4 waves).
// Each wave owns a 16-row q tile; streams K/V in 32-key slabs:
//   scores(16x32)  : 4 WMMA  (Q 16x64 . K^T 64x32)
//   online softmax : shfl_xor row-reductions within 16-lane halves
//   P.V (16x64)    : 4 WMMA  (P via LDS C->A layout round-trip, V^T B-frags)
// ---------------------------------------------------------------------
__global__ __launch_bounds__(128) void attn_wmma(
    const _Float16* __restrict__ Q, const _Float16* __restrict__ Kt,
    const _Float16* __restrict__ VT, const float* __restrict__ graph,
    const unsigned char* __restrict__ mask, _Float16* __restrict__ O,
    int use_graph)
{
  const int qt   = blockIdx.x & (S_ / 64 - 1);
  const int h    = (blockIdx.x / (S_ / 64)) & (NH_ - 1);
  const int b    = blockIdx.x / ((S_ / 64) * NH_);
  const int wave = threadIdx.x >> 5;
  const int lane = threadIdx.x & 31;
  const int ln   = lane & 15;
  const int hf   = lane >> 4;
  const int q0   = qt * 64 + wave * 16;

  __shared__ __align__(16) _Float16 psh[4][16 * 32];
  _Float16* pw = &psh[wave][0];

  const _Float16* Qbase = Q + ((size_t)(b * S_ + q0)) * H_ + h * DH_;
  v16h qf0 = load_frag(Qbase,      H_);
  v16h qf1 = load_frag(Qbase + 32, H_);

  float m_r[8], l_r[8];
  v8f o_acc[4];
#pragma unroll
  for (int r = 0; r < 8; ++r) { m_r[r] = -3.0e38f; l_r[r] = 0.0f; }
#pragma unroll
  for (int dc = 0; dc < 4; ++dc) o_acc[dc] = (v8f){};

  const float rsc = 0.125f;  // 1/sqrt(64)

  for (int k0 = 0; k0 < S_; k0 += 32) {
    const _Float16* Kb0 = Kt + ((size_t)(b * S_ + k0))      * H_ + h * DH_;
    const _Float16* Kb1 = Kt + ((size_t)(b * S_ + k0 + 16)) * H_ + h * DH_;
    v8f s0 = {}, s1 = {};
    s0 = __builtin_amdgcn_wmma_f32_16x16x32_f16(false, qf0, false, load_frag(Kb0,      H_), (short)0, s0, false, false);
    s0 = __builtin_amdgcn_wmma_f32_16x16x32_f16(false, qf1, false, load_frag(Kb0 + 32, H_), (short)0, s0, false, false);
    s1 = __builtin_amdgcn_wmma_f32_16x16x32_f16(false, qf0, false, load_frag(Kb1,      H_), (short)0, s1, false, false);
    s1 = __builtin_amdgcn_wmma_f32_16x16x32_f16(false, qf1, false, load_frag(Kb1 + 32, H_), (short)0, s1, false, false);

    const int kc0 = k0 + ln, kc1 = k0 + 16 + ln;
    const unsigned char mk0 = mask[b * S_ + kc0];
    const unsigned char mk1 = mask[b * S_ + kc1];

#pragma unroll
    for (int r = 0; r < 8; ++r) {
      float sv0 = s0[r] * rsc;
      float sv1 = s1[r] * rsc;
      const int qrow = q0 + r + 8 * hf;
      if (use_graph && qrow < 100) {   // scores * (1 - graph) on top-left block
        if (kc0 < 100) sv0 *= (1.0f - graph[(size_t)b * 10000 + qrow * 100 + kc0]);
        if (kc1 < 100) sv1 *= (1.0f - graph[(size_t)b * 10000 + qrow * 100 + kc1]);
      }
      if (mk0) sv0 = -1.0e9f;
      if (mk1) sv1 = -1.0e9f;

      float mx = fmaxf(sv0, sv1);
#pragma unroll
      for (int o = 1; o < 16; o <<= 1) mx = fmaxf(mx, __shfl_xor(mx, o, 32));
      const float mn = fmaxf(m_r[r], mx);
      const float sc = __expf(m_r[r] - mn);
      const float e0 = __expf(sv0 - mn);
      const float e1 = __expf(sv1 - mn);
      float sum = e0 + e1;
#pragma unroll
      for (int o = 1; o < 16; o <<= 1) sum += __shfl_xor(sum, o, 32);
      l_r[r] = l_r[r] * sc + sum;
      m_r[r] = mn;
#pragma unroll
      for (int dc = 0; dc < 4; ++dc) o_acc[dc][r] *= sc;

      const int rr = r + 8 * hf;                 // C-layout -> row-major P in LDS
      pw[rr * 32 + ln]      = (_Float16)e0;
      pw[rr * 32 + 16 + ln] = (_Float16)e1;
    }

    v16h pa = load_frag(pw, 32);
#pragma unroll
    for (int dc = 0; dc < 4; ++dc) {
      const _Float16* Vb = VT + (((size_t)(b * NH_ + h)) * DH_ + dc * 16) * S_ + k0;
      o_acc[dc] = __builtin_amdgcn_wmma_f32_16x16x32_f16(
          false, pa, false, load_frag(Vb, S_), (short)0, o_acc[dc], false, false);
    }
  }

#pragma unroll
  for (int dc = 0; dc < 4; ++dc) {
#pragma unroll
    for (int r = 0; r < 8; ++r) {
      const float v = o_acc[dc][r] / l_r[r];
      O[((size_t)(b * S_ + q0 + r + 8 * hf)) * H_ + h * DH_ + dc * 16 + ln] = (_Float16)v;
    }
  }
}

// ---------------------------------------------------------------------
// Residual + LayerNorm: out = LN(res + add)*g + b ; writes f32 and/or f16.
// ---------------------------------------------------------------------
__global__ __launch_bounds__(256) void ln_kernel(
    const float* __restrict__ res, const float* __restrict__ add,
    const float* __restrict__ g, const float* __restrict__ beta,
    float* of, _Float16* oh)
{
  const int row = blockIdx.x;
  const int t   = threadIdx.x;
  const float* rp = res + (size_t)row * H_;
  const float* ap = add + (size_t)row * H_;
  __shared__ float sred[8];

  float v0 = rp[t] + ap[t];
  float v1 = rp[t + 256] + ap[t + 256];

  float s = v0 + v1;
#pragma unroll
  for (int o = 1; o < 32; o <<= 1) s += __shfl_xor(s, o, 32);
  if ((t & 31) == 0) sred[t >> 5] = s;
  __syncthreads();
  float tot = 0.0f;
#pragma unroll
  for (int i = 0; i < 8; ++i) tot += sred[i];
  const float mean = tot * (1.0f / (float)H_);
  __syncthreads();

  const float d0 = v0 - mean, d1 = v1 - mean;
  float s2 = d0 * d0 + d1 * d1;
#pragma unroll
  for (int o = 1; o < 32; o <<= 1) s2 += __shfl_xor(s2, o, 32);
  if ((t & 31) == 0) sred[t >> 5] = s2;
  __syncthreads();
  float tv = 0.0f;
#pragma unroll
  for (int i = 0; i < 8; ++i) tv += sred[i];
  const float inv = rsqrtf(tv * (1.0f / (float)H_) + 1e-6f);

  const float o0 = d0 * inv * g[t] + beta[t];
  const float o1 = d1 * inv * g[t + 256] + beta[t + 256];
  if (of) { of[(size_t)row * H_ + t] = o0; of[(size_t)row * H_ + t + 256] = o1; }
  if (oh) { oh[(size_t)row * H_ + t] = (_Float16)o0; oh[(size_t)row * H_ + t + 256] = (_Float16)o1; }
}

// ---- weight convert+transpose: W[K,N] f32 -> WT[N,K] f16 ----
__global__ void wconv_kernel(const float* __restrict__ w, _Float16* __restrict__ wt,
                             int K, int N)
{
  const int i = blockIdx.x * 256 + threadIdx.x;
  if (i < K * N) {
    const int k = i / N, n = i % N;
    wt[(size_t)n * K + k] = (_Float16)w[i];
  }
}

// ---- f32 -> f16 elementwise ----
__global__ void h16_kernel(const float* __restrict__ x, _Float16* __restrict__ o, int n)
{
  const int i = blockIdx.x * 256 + threadIdx.x;
  if (i < n) o[i] = (_Float16)x[i];
}

// ---- V[B,S,H] f16 -> VT[B,NH,DH,S] f16 (per-head transpose) ----
__global__ void vtrans_kernel(const _Float16* __restrict__ V, _Float16* __restrict__ VT)
{
  const size_t i = (size_t)blockIdx.x * 256 + threadIdx.x;
  if (i < GE_) {
    const int col = (int)(i % H_);
    const size_t row = i / H_;
    const int b = (int)(row / S_), s = (int)(row % S_);
    const int h = col / DH_, d = col % DH_;
    VT[(((size_t)b * NH_ + h) * DH_ + d) * S_ + s] = V[i];
  }
}

// ---------------------------------------------------------------------
extern "C" void kernel_launch(void* const* d_in, const int* in_sizes, int n_in,
                              void* d_out, int out_size, void* d_ws, size_t ws_size,
                              hipStream_t stream)
{
  const float* x      = (const float*)d_in[0];
  const float* y      = (const float*)d_in[1];
  const unsigned char* xmask = (const unsigned char*)d_in[2];
  const unsigned char* ymask = (const unsigned char*)d_in[3];
  const float* graphx = (const float*)d_in[4];
  const float* a1_wq = (const float*)d_in[5];  const float* a1_bq = (const float*)d_in[6];
  const float* a1_wk = (const float*)d_in[7];  const float* a1_bk = (const float*)d_in[8];
  const float* a1_wv = (const float*)d_in[9];  const float* a1_bv = (const float*)d_in[10];
  const float* a1_wm = (const float*)d_in[11]; const float* a1_bm = (const float*)d_in[12];
  const float* a2_wq = (const float*)d_in[13]; const float* a2_bq = (const float*)d_in[14];
  const float* a2_wk = (const float*)d_in[15]; const float* a2_bk = (const float*)d_in[16];
  const float* a2_wv = (const float*)d_in[17]; const float* a2_bv = (const float*)d_in[18];
  const float* a2_wm = (const float*)d_in[19]; const float* a2_bm = (const float*)d_in[20];
  const float* ffn_w1 = (const float*)d_in[21]; const float* ffn_b1 = (const float*)d_in[22];
  const float* ffn_w2 = (const float*)d_in[23]; const float* ffn_b2 = (const float*)d_in[24];
  const float* ln1_g = (const float*)d_in[25]; const float* ln1_b = (const float*)d_in[26];
  const float* ln2_g = (const float*)d_in[27]; const float* ln2_b = (const float*)d_in[28];
  const float* ln3_g = (const float*)d_in[29]; const float* ln3_b = (const float*)d_in[30];

  // ---------------- workspace carve-up (128 MB total) ----------------
  char* ws = (char*)d_ws;
  size_t off = 0;
  auto a16 = [&](size_t n) { _Float16* p = (_Float16*)(ws + off); off += n * sizeof(_Float16); return p; };
  _Float16* W16   = a16(GE_);   // 8 * HxH + 2 * HxFF transposed f16 weights
  _Float16* xh    = a16(GE_);
  _Float16* yh    = a16(GE_);
  _Float16* x1h   = a16(GE_);
  _Float16* x2h   = a16(GE_);
  _Float16* attnO = a16(GE_);
  _Float16* qf    = a16(GE_);   // qf..VT (4*GE_ f16 = 32MB) aliased as FFN hidden
  _Float16* kf    = a16(GE_);
  _Float16* vf    = a16(GE_);
  _Float16* VT    = a16(GE_);
  _Float16* hbuf  = qf;         // MS_ x FF_ f16 == 4*GE_ f16
  auto a32 = [&](size_t n) { float* p = (float*)(ws + off); off += n * sizeof(float); return p; };
  float* gemmO = a32(GE_);
  float* x1    = a32(GE_);
  float* x2    = a32(GE_);

  const size_t WHH = (size_t)H_ * H_;
  _Float16* tw[8] = { W16, W16 + WHH, W16 + 2*WHH, W16 + 3*WHH,
                      W16 + 4*WHH, W16 + 5*WHH, W16 + 6*WHH, W16 + 7*WHH };
  _Float16* w1T = W16 + 8 * WHH;            // [FF_, H_]
  _Float16* w2T = w1T + (size_t)H_ * FF_;   // [H_, FF_]

  auto wconv = [&](const float* w, _Float16* dst, int K, int N) {
    const int n = K * N;
    wconv_kernel<<<(n + 255) / 256, 256, 0, stream>>>(w, dst, K, N);
  };
  auto gemm = [&](const _Float16* A, const _Float16* WTp, const float* bias,
                  float* Cf, _Float16* Ch, int M, int N, int K, int relu) {
    gemm_wmma<<<dim3(M / 32, N / 64), 256, 0, stream>>>(A, WTp, bias, Cf, Ch, M, N, K, relu);
  };

  // ---------------- weight / input conversion ----------------
  wconv(a1_wq, tw[0], H_, H_); wconv(a1_wk, tw[1], H_, H_);
  wconv(a1_wv, tw[2], H_, H_); wconv(a1_wm, tw[3], H_, H_);
  wconv(a2_wq, tw[4], H_, H_); wconv(a2_wk, tw[5], H_, H_);
  wconv(a2_wv, tw[6], H_, H_); wconv(a2_wm, tw[7], H_, H_);
  wconv(ffn_w1, w1T, H_, FF_); wconv(ffn_w2, w2T, FF_, H_);
  h16_kernel<<<(int)(GE_ / 256), 256, 0, stream>>>(x, xh, (int)GE_);
  h16_kernel<<<(int)(GE_ / 256), 256, 0, stream>>>(y, yh, (int)GE_);

  const int attn_blocks = B_ * NH_ * (S_ / 64);

  // ---------------- stage 1: graph-gated self-attention ----------------
  gemm(xh, tw[0], a1_bq, nullptr, qf, MS_, H_, H_, 0);
  gemm(xh, tw[1], a1_bk, nullptr, kf, MS_, H_, H_, 0);
  gemm(xh, tw[2], a1_bv, nullptr, vf, MS_, H_, H_, 0);
  vtrans_kernel<<<(int)(GE_ / 256), 256, 0, stream>>>(vf, VT);
  attn_wmma<<<attn_blocks, 128, 0, stream>>>(qf, kf, VT, graphx, xmask, attnO, 1);
  gemm(attnO, tw[3], a1_bm, gemmO, nullptr, MS_, H_, H_, 0);
  ln_kernel<<<MS_, 256, 0, stream>>>(x, gemmO, ln1_g, ln1_b, x1, x1h);

  // ---------------- stage 2: cross-attention (q from x, k/v from y) ----
  gemm(x1h, tw[4], a2_bq, nullptr, qf, MS_, H_, H_, 0);
  gemm(yh,  tw[5], a2_bk, nullptr, kf, MS_, H_, H_, 0);
  gemm(yh,  tw[6], a2_bv, nullptr, vf, MS_, H_, H_, 0);
  vtrans_kernel<<<(int)(GE_ / 256), 256, 0, stream>>>(vf, VT);
  attn_wmma<<<attn_blocks, 128, 0, stream>>>(qf, kf, VT, nullptr, ymask, attnO, 0);
  gemm(attnO, tw[7], a2_bm, gemmO, nullptr, MS_, H_, H_, 0);
  ln_kernel<<<MS_, 256, 0, stream>>>(x1, gemmO, ln2_g, ln2_b, x2, x2h);

  // ---------------- stage 3: FFN ----------------
  gemm(x2h, w1T, ffn_b1, nullptr, hbuf, MS_, FF_, H_, 1);       // ReLU
  gemm(hbuf, w2T, ffn_b2, gemmO, nullptr, MS_, H_, FF_, 0);
  ln_kernel<<<MS_, 256, 0, stream>>>(x2, gemmO, ln3_g, ln3_b, (float*)d_out, nullptr);

  (void)in_sizes; (void)n_in; (void)out_size; (void)ws_size;
}
